// REINFORCE_DM_7069516169861
// MI455X (gfx1250) — matrix operand order repaired and checked by reading, not compiled, not run on hardware
//
#include <hip/hip_runtime.h>
#include <hip/hip_bf16.h>

typedef __attribute__((ext_vector_type(16))) __bf16 v16bf;
typedef __attribute__((ext_vector_type(8)))  __bf16 v8bf;
typedef __attribute__((ext_vector_type(8)))  float  v8f;
typedef __attribute__((ext_vector_type(4)))  float  v4f;

#define HD     128      // hidden dim
#define KF     256      // concat embedding K (2*H), pt handled separately
#define FPAD   264      // padded feats row (bf16) -> conflict-free LDS b128
#define H0PAD  136      // padded h0 row (bf16)
#define W0PAD  264      // padded W0T row in LDS
#define W1PAD  136      // padded W1T row in LDS
#define WPB    4        // waves per block
#define RPW    16       // rows per wave (one 16xK A tile)
#define RPB    (WPB*RPW)

// ---------------- weight prep: f32 -> n-major bf16 ----------------
__global__ void rdm_prep(const float* __restrict__ W0, const float* __restrict__ W1,
                         __bf16* __restrict__ W0T, __bf16* __restrict__ W1T) {
    int tid = blockIdx.x * blockDim.x + threadIdx.x;
    int stride = gridDim.x * blockDim.x;
    for (int i = tid; i < HD * KF; i += stride) {      // W0T[n][k] = W0[k][n], k<256
        int n = i >> 8, k = i & 255;
        W0T[i] = (__bf16)W0[k * HD + n];
    }
    for (int i = tid; i < HD * HD; i += stride) {      // W1T[n][k] = W1[k][n]
        int n = i >> 7, k = i & 127;
        W1T[i] = (__bf16)W1[k * HD + n];
    }
}

// A fragment (16x32 bf16) per ISA layout from a row-major [16][ldk] LDS tile.
// lanes 0-15: row=lane, K={0..7,16..23}; lanes 16-31: row=lane-16, K={8..15,24..31}
static __device__ inline v16bf rdm_load_a(const __bf16* base, int ldk, int kc, int lane) {
    const int m = lane & 15;
    const int half = lane >> 4;
    const __bf16* p = base + m * ldk + (kc << 5) + half * 8;
    v8bf lo = *(const v8bf*)(p);
    v8bf hi = *(const v8bf*)(p + 16);
    return __builtin_shufflevector(lo, hi, 0,1,2,3,4,5,6,7,8,9,10,11,12,13,14,15);
}

// B fragment (32x16 bf16) from LDS-staged n-major weights [HD][ldk]:
// lane holds 16 K at fixed N = nt*16 + lane%16;
// lanes 0-15 K=kc*32+{0..7,16..23}, lanes 16-31 K=kc*32+{8..15,24..31}.
static __device__ inline v16bf rdm_load_b(const __bf16* wT, int ldk,
                                          int nt, int kc, int lane) {
    const int n  = (nt << 4) + (lane & 15);
    const int k0 = (kc << 5) + ((lane >> 4) << 3);
    const __bf16* p = wT + n * ldk + k0;
    v8bf lo = *(const v8bf*)(p);
    v8bf hi = *(const v8bf*)(p + 16);
    return __builtin_shufflevector(lo, hi, 0,1,2,3,4,5,6,7,8,9,10,11,12,13,14,15);
}

// ---------------- fused gather + 3-layer MLP (bf16 WMMA, f32 accum) ----------------
__global__ __launch_bounds__(WPB * 32)
void rdm_score(const float* __restrict__ m_emb, const float* __restrict__ op_emb,
               const float* __restrict__ proc,  const int* __restrict__ m_ids,
               const int* __restrict__ op_idxs,
               const __bf16* __restrict__ W0T, const float* __restrict__ w0_pt,
               const __bf16* __restrict__ W1T,
               const float* __restrict__ b0, const float* __restrict__ b1,
               const float* __restrict__ w2, const float* __restrict__ b2,
               const int* __restrict__ mpt, float* __restrict__ scores, int N) {
    __shared__ __bf16 s_w0[HD][W0PAD];           // staged W0T, bf16  (~67.6KB)
    __shared__ __bf16 s_w1[HD][W1PAD];           // staged W1T, bf16  (~34.8KB)
    __shared__ __bf16 s_feats[WPB][RPW][FPAD];   // gathered features (~33KB)
    __shared__ __bf16 s_h0[WPB][RPW][H0PAD];     // layer-0 activations (~17KB)

    const int lane = threadIdx.x & 31;
    const int wave = threadIdx.x >> 5;
    const int rowBase = blockIdx.x * RPB + wave * RPW;
    const float inv_mpt = 1.0f / (float)mpt[0];

    // ---- stage gathered embeddings into LDS as bf16 (coalesced 16B loads) ----
    // deep unroll: keep many irregular gathers in flight (HBM is the roofline)
    #pragma unroll 8
    for (int r = 0; r < RPW; ++r) {
        int row = rowBase + r; if (row >= N) row = N - 1;
        const float* msrc = m_emb  + (size_t)m_ids[row]   * HD;
        const float* osrc = op_emb + (size_t)op_idxs[row] * HD;
        v4f mv = *(const v4f*)(msrc + lane * 4);
        v4f ov = *(const v4f*)(osrc + lane * 4);
        __bf16* dst = &s_feats[wave][r][0];
        #pragma unroll
        for (int i = 0; i < 4; ++i) {
            dst[lane * 4 + i]      = (__bf16)mv[i];
            dst[HD + lane * 4 + i] = (__bf16)ov[i];
        }
    }

    // ---- cooperative stage of bf16 weights into padded LDS (once per block) ----
    // W0T: 128 rows x 32 chunks of 8 bf16 (16B); W1T: 128 rows x 16 chunks
    for (int c = threadIdx.x; c < HD * 32; c += WPB * 32) {
        int n = c >> 5, ck = c & 31;
        *(v4f*)&s_w0[n][ck * 8] = *(const v4f*)(W0T + n * KF + ck * 8);
    }
    for (int c = threadIdx.x; c < HD * 16; c += WPB * 32) {
        int n = c >> 4, ck = c & 15;
        *(v4f*)&s_w1[n][ck * 8] = *(const v4f*)(W1T + n * HD + ck * 8);
    }
    __syncthreads();   // weights are block-shared

    const int half = lane >> 4;
    const int col  = lane & 15;

    float ptv[8];
    #pragma unroll
    for (int r = 0; r < 8; ++r) {
        int row = rowBase + r + 8 * half; if (row >= N) row = N - 1;
        ptv[r] = proc[row] * inv_mpt;
    }

    // ---- layer 0: h0[16,128] = feats[16,256] @ W0[256,128] ----
    // nt split into 2 halves of 4 tiles -> only 32 live accumulator VGPRs
    #pragma unroll
    for (int ntH = 0; ntH < 2; ++ntH) {
        v8f acc[4];
        #pragma unroll
        for (int q = 0; q < 4; ++q) acc[q] = (v8f){0.f,0.f,0.f,0.f,0.f,0.f,0.f,0.f};
        #pragma unroll
        for (int kc = 0; kc < 8; ++kc) {
            v16bf a = rdm_load_a(&s_feats[wave][0][0], FPAD, kc, lane);
            #pragma unroll
            for (int q = 0; q < 4; ++q) {
                v16bf b = rdm_load_b(&s_w0[0][0], W0PAD, ntH * 4 + q, kc, lane);
                acc[q] = __builtin_amdgcn_wmma_f32_16x16x32_bf16(
                    false, a, false, b, (short)0, acc[q], false, false);
            }
        }
        // epilogue: + b0 + pt*W0[256,:], leaky_relu(0.01), -> bf16 LDS
        #pragma unroll
        for (int q = 0; q < 4; ++q) {
            const int n = ((ntH * 4 + q) << 4) + col;
            const float bb = b0[n], wpt = w0_pt[n];
            #pragma unroll
            for (int r = 0; r < 8; ++r) {
                float x = acc[q][r] + bb + ptv[r] * wpt;
                x = x > 0.f ? x : 0.01f * x;
                s_h0[wave][r + 8 * half][n] = (__bf16)x;
            }
        }
    }

    // ---- layer 1 + layer 2 fused: keep h1 in registers, dot with W2 ----
    float partial[8];
    #pragma unroll
    for (int r = 0; r < 8; ++r) partial[r] = 0.f;

    #pragma unroll
    for (int ntH = 0; ntH < 2; ++ntH) {
        v8f acc[4];
        #pragma unroll
        for (int q = 0; q < 4; ++q) acc[q] = (v8f){0.f,0.f,0.f,0.f,0.f,0.f,0.f,0.f};
        #pragma unroll
        for (int kc = 0; kc < 4; ++kc) {
            v16bf a = rdm_load_a(&s_h0[wave][0][0], H0PAD, kc, lane);
            #pragma unroll
            for (int q = 0; q < 4; ++q) {
                v16bf b = rdm_load_b(&s_w1[0][0], W1PAD, ntH * 4 + q, kc, lane);
                acc[q] = __builtin_amdgcn_wmma_f32_16x16x32_bf16(
                    false, a, false, b, (short)0, acc[q], false, false);
            }
        }
        // h1 = leaky(acc + b1); partial[r] += h1 * w2[n]  (no LDS round-trip)
        #pragma unroll
        for (int q = 0; q < 4; ++q) {
            const int n = ((ntH * 4 + q) << 4) + col;
            const float bb = b1[n], wv = w2[n];
            #pragma unroll
            for (int r = 0; r < 8; ++r) {
                float x = acc[q][r] + bb;
                x = x > 0.f ? x : 0.01f * x;
                partial[r] += x * wv;
            }
        }
    }

    // butterfly-reduce each row's dot across the 16 lanes of this half
    #pragma unroll
    for (int off = 1; off < 16; off <<= 1) {
        #pragma unroll
        for (int r = 0; r < 8; ++r) partial[r] += __shfl_xor(partial[r], off, 32);
    }
    if (col == 0) {
        const float bias2 = b2[0];
        #pragma unroll
        for (int r = 0; r < 8; ++r) {
            int row = rowBase + r + 8 * half;
            if (row < N) scores[row] = partial[r] + bias2;
        }
    }
}

// ---------------- softmax reductions ----------------
__global__ void rdm_pmax(const float* __restrict__ s, float* __restrict__ partial, int N) {
    __shared__ float red[256];
    float mx = -__builtin_inff();
    for (int i = blockIdx.x * blockDim.x + threadIdx.x; i < N; i += gridDim.x * blockDim.x)
        mx = fmaxf(mx, s[i]);
    red[threadIdx.x] = mx; __syncthreads();
    for (int o = 128; o > 0; o >>= 1) {
        if (threadIdx.x < o) red[threadIdx.x] = fmaxf(red[threadIdx.x], red[threadIdx.x + o]);
        __syncthreads();
    }
    if (threadIdx.x == 0) partial[blockIdx.x] = red[0];
}

__global__ void rdm_fmax(const float* __restrict__ partial, float* __restrict__ gstats) {
    __shared__ float red[512];
    red[threadIdx.x] = partial[threadIdx.x]; __syncthreads();
    for (int o = 256; o > 0; o >>= 1) {
        if (threadIdx.x < o) red[threadIdx.x] = fmaxf(red[threadIdx.x], red[threadIdx.x + o]);
        __syncthreads();
    }
    if (threadIdx.x == 0) gstats[0] = red[0];
}

__global__ void rdm_exp(const float* __restrict__ s, const float* __restrict__ gstats,
                        float* __restrict__ probs, float* __restrict__ partial, int N) {
    __shared__ float red[256];
    const float gmax = gstats[0];
    float sum = 0.f;
    for (int i = blockIdx.x * blockDim.x + threadIdx.x; i < N; i += gridDim.x * blockDim.x) {
        float e = __expf(s[i] - gmax);
        probs[i] = e;
        sum += e;
    }
    red[threadIdx.x] = sum; __syncthreads();
    for (int o = 128; o > 0; o >>= 1) {
        if (threadIdx.x < o) red[threadIdx.x] += red[threadIdx.x + o];
        __syncthreads();
    }
    if (threadIdx.x == 0) partial[blockIdx.x] = red[0];
}

__global__ void rdm_fsum(const float* __restrict__ partial, float* __restrict__ gstats) {
    __shared__ float red[512];
    red[threadIdx.x] = partial[threadIdx.x]; __syncthreads();
    for (int o = 256; o > 0; o >>= 1) {
        if (threadIdx.x < o) red[threadIdx.x] += red[threadIdx.x + o];
        __syncthreads();
    }
    if (threadIdx.x == 0) { gstats[1] = red[0]; gstats[2] = 1.0f / red[0]; }
}

__global__ void rdm_scale(float* __restrict__ out, const float* __restrict__ gstats, int N) {
    int i = blockIdx.x * blockDim.x + threadIdx.x;
    if (i < N)            out[i] *= gstats[2];
    else if (i < N + 1000) out[i] = -1.0f;     // selected default
    else if (i == N + 1000) out[i] = 0.0f;     // count default
}

// ---------------- greedy NMS selection (single persistent workgroup) ----------------
#define NSEG_MAX 512
__global__ __launch_bounds__(1024)
void rdm_select(const float* __restrict__ probs, const int* __restrict__ job_ids,
                const int* __restrict__ m_ids, float* __restrict__ out_sel,
                float* __restrict__ out_cnt, int N) {
    __shared__ unsigned jobK[(5000 + 31) / 32];
    __shared__ unsigned machK[(1000 + 31) / 32];
    __shared__ float segVal[NSEG_MAX];
    __shared__ int   segIdx[NSEG_MAX];
    __shared__ int   segJob[NSEG_MAX];
    __shared__ int   segMach[NSEG_MAX];
    __shared__ float rv[1024];
    __shared__ int   ri[1024];
    __shared__ int   sJ, sM, sStop;

    const int tid = threadIdx.x, lane = tid & 31, wave = tid >> 5;
    int segSize = 1024;
    while (segSize * NSEG_MAX < N) segSize <<= 1;
    const int NSEG = (N + segSize - 1) / segSize;

    for (int i = tid; i < (5000 + 31) / 32; i += 1024) jobK[i] = 0u;
    for (int i = tid; i < (1000 + 31) / 32; i += 1024) machK[i] = 0u;
    __syncthreads();

    // wave-cooperative segment champion scan (s is wave-uniform)
    auto scan_seg = [&](int s) {
        const int base = s * segSize;
        int end = base + segSize; if (end > N) end = N;
        float bv = -__builtin_inff(); int bi = -1;
        for (int i = base + lane; i < end; i += 32) {
            float p = probs[i];
            if (p > 0.f) {
                int j = job_ids[i], mm = m_ids[i];
                if (!((jobK[j >> 5] >> (j & 31)) & 1u) && !((machK[mm >> 5] >> (mm & 31)) & 1u)) {
                    if (p > bv || (p == bv && i < bi)) { bv = p; bi = i; }
                }
            }
        }
        #pragma unroll
        for (int off = 16; off > 0; off >>= 1) {
            float ov = __shfl_xor(bv, off, 32);
            int   oi = __shfl_xor(bi, off, 32);
            if (ov > bv || (ov == bv && oi != -1 && (bi == -1 || oi < bi))) { bv = ov; bi = oi; }
        }
        if (lane == 0) {
            segVal[s] = bv; segIdx[s] = bi;
            segJob[s]  = (bi >= 0) ? job_ids[bi] : -1;
            segMach[s] = (bi >= 0) ? m_ids[bi]   : -1;
        }
    };

    for (int s = wave; s < NSEG; s += 32) scan_seg(s);
    __syncthreads();

    int count = 0;
    for (int it = 0; it < 1000; ++it) {
        float bv = -__builtin_inff(); int bi = -1;
        if (tid < NSEG) { bv = segVal[tid]; bi = segIdx[tid]; }
        rv[tid] = bv; ri[tid] = bi; __syncthreads();
        for (int o = 512; o > 0; o >>= 1) {
            if (tid < o) {
                float ov = rv[tid + o]; int oi = ri[tid + o];
                if (ov > rv[tid] || (ov == rv[tid] && oi != -1 && (ri[tid] == -1 || oi < ri[tid]))) {
                    rv[tid] = ov; ri[tid] = oi;
                }
            }
            __syncthreads();
        }
        if (tid == 0) {
            int bidx = ri[0];
            if (bidx < 0 || rv[0] <= 0.f) sStop = 1;
            else {
                sStop = 0;
                out_sel[count] = (float)bidx;
                int J = job_ids[bidx], M = m_ids[bidx];
                jobK[J >> 5]  |= 1u << (J & 31);
                machK[M >> 5] |= 1u << (M & 31);
                sJ = J; sM = M;
            }
        }
        __syncthreads();
        if (sStop) break;
        ++count;
        const int J = sJ, M = sM;
        for (int s = wave; s < NSEG; s += 32)
            if (segIdx[s] >= 0 && (segJob[s] == J || segMach[s] == M)) scan_seg(s);
        __syncthreads();
    }
    if (tid == 0) *out_cnt = (float)count;
}

// ---------------- launcher ----------------
extern "C" void kernel_launch(void* const* d_in, const int* in_sizes, int n_in,
                              void* d_out, int out_size, void* d_ws, size_t ws_size,
                              hipStream_t stream) {
    const float* m_emb  = (const float*)d_in[0];
    const float* op_emb = (const float*)d_in[1];
    const float* proc   = (const float*)d_in[2];
    const int*   m_ids  = (const int*)d_in[3];
    const int*   op_idx = (const int*)d_in[4];
    const int*   job_id = (const int*)d_in[5];
    const float* W0     = (const float*)d_in[6];
    const float* b0     = (const float*)d_in[7];
    const float* W1     = (const float*)d_in[8];
    const float* b1     = (const float*)d_in[9];
    const float* W2     = (const float*)d_in[10];
    const float* b2     = (const float*)d_in[11];
    const int*   mpt    = (const int*)d_in[12];
    const int N = in_sizes[2];

    char* ws = (char*)d_ws;
    size_t off = 0;
    __bf16* W0T   = (__bf16*)(ws + off); off += (size_t)HD * KF * 2;
    __bf16* W1T   = (__bf16*)(ws + off); off += (size_t)HD * HD * 2;
    float* gstats = (float*)(ws + off);  off += 64;
    float* partial= (float*)(ws + off);  off += 512 * 4;
    float* scores = (float*)(ws + off);  off += (size_t)N * 4;

    float* probs   = (float*)d_out;
    float* out_sel = probs + N;
    float* out_cnt = probs + N + 1000;

    rdm_prep<<<64, 256, 0, stream>>>(W0, W1, W0T, W1T);
    rdm_score<<<(N + RPB - 1) / RPB, WPB * 32, 0, stream>>>(
        m_emb, op_emb, proc, m_ids, op_idx,
        W0T, W0 + 256 * HD, W1T, b0, b1, W2, b2, mpt, scores, N);
    rdm_pmax<<<512, 256, 0, stream>>>(scores, partial, N);
    rdm_fmax<<<1, 512, 0, stream>>>(partial, gstats);
    rdm_exp<<<512, 256, 0, stream>>>(scores, gstats, probs, partial, N);
    rdm_fsum<<<1, 512, 0, stream>>>(partial, gstats);
    rdm_scale<<<(N + 1001 + 255) / 256, 256, 0, stream>>>(probs, gstats, N);
    rdm_select<<<1, 1024, 0, stream>>>(probs, job_id, m_ids, out_sel, out_cnt, N);
}
